// AttentionHead_71219147702834
// MI455X (gfx1250) — matrix operand order repaired and checked by reading, not compiled, not run on hardware
//
#include <hip/hip_runtime.h>
#include <hip/hip_bf16.h>

// Problem constants (match the reference)
#define BATCH 8
#define SEQ   4096
#define EMB   768
#define HD    64

typedef __attribute__((ext_vector_type(16))) __bf16 v16bf;
typedef __attribute__((ext_vector_type(8)))  float  v8f;

union FragAB { unsigned u[8]; v16bf v; };

// f32x2 -> packed bf16x2, round-half-up (1 ulp-tie difference vs RNE).
__device__ __forceinline__ unsigned f2bf2(float lo, float hi) {
  unsigned a = __builtin_bit_cast(unsigned, lo) + 0x8000u;
  unsigned b = __builtin_bit_cast(unsigned, hi) + 0x8000u;
#if __has_builtin(__builtin_amdgcn_perm)
  return __builtin_amdgcn_perm(b, a, 0x07060302u);  // one v_perm_b32
#else
  return (a >> 16) | (b & 0xFFFF0000u);
#endif
}
__device__ __forceinline__ unsigned short f2bf1(float x) {
  unsigned a = __builtin_bit_cast(unsigned, x) + 0x8000u;
  return (unsigned short)(a >> 16);
}

__device__ __forceinline__ v8f wmma_bf16(const FragAB& a, const FragAB& b, v8f c) {
  // D = A(16x32 bf16) * B(32x16 bf16) + C(16x16 f32)
  return __builtin_amdgcn_wmma_f32_16x16x32_bf16(false, a.v, false, b.v,
                                                 (short)0, c, false, false);
}

// ---------------------------------------------------------------------------
// Kernel 0: Wq/Wk f32 -> bf16 (tiny, once)
// ---------------------------------------------------------------------------
__global__ void wconvert(const float* __restrict__ wq, const float* __restrict__ wk,
                         unsigned short* __restrict__ wq_bf,
                         unsigned short* __restrict__ wk_bf) {
  int i = blockIdx.x * 256 + threadIdx.x;
  if (i < HD * EMB) {
    wq_bf[i] = f2bf1(wq[i]);
    wk_bf[i] = f2bf1(wk[i]);
  }
}

// ---------------------------------------------------------------------------
// Kernel 1: q = x @ Wq^T, k = x @ Wk^T  (M=B*S=32768, N=64, K=768), bf16 out.
// Also stores q transposed (qT[b][h][s]) so kernel 3's V^T A-fragments become
// contiguous b128 loads. One wave per 16-row M tile; 8 waves per block.
// ---------------------------------------------------------------------------
__global__ __launch_bounds__(256) void proj_qk(
    const float* __restrict__ x,
    const unsigned short* __restrict__ wq_bf,
    const unsigned short* __restrict__ wk_bf,
    unsigned short* __restrict__ q_bf,
    unsigned short* __restrict__ k_bf,
    unsigned short* __restrict__ vT_bf)
{
  const int lane = threadIdx.x & 31;
  const int wv   = threadIdx.x >> 5;
  const int tile = blockIdx.x * 8 + wv;      // 0..2047 M-tiles
  const int n  = lane & 15;
  const int hf = lane >> 4;
  const long rowBase = (long)tile * 16;
  const int  b    = tile >> 8;               // 256 tiles per batch
  const int  srow = (tile & 255) * 16;       // seq row within batch

  v8f accQ[4], accK[4];
#pragma unroll
  for (int t = 0; t < 4; ++t) { accQ[t] = (v8f){}; accK[t] = (v8f){}; }

  for (int kc = 0; kc < EMB; kc += 32) {
    // A fragment: x rows, converted f32 -> bf16 on the fly
    FragAB a;
    const long xrow = (rowBase + n) * EMB;
#pragma unroll
    for (int v = 0; v < 8; ++v) {
      int kb = ((v < 4) ? 0 : 16) + hf * 8 + (v & 3) * 2;   // A-layout K pair
      const float2 xv = *(const float2*)(x + xrow + kc + kb);
      a.u[v] = f2bf2(xv.x, xv.y);
    }
#pragma unroll
    for (int nt = 0; nt < 4; ++nt) {                         // H = 64 -> 4 N tiles
      FragAB bq, bk;
      const int h = nt * 16 + n;
#pragma unroll
      for (int v = 0; v < 8; ++v) {
        int e = kc + ((v < 4) ? 0 : 16) + hf * 8 + (v & 3) * 2; // B-layout K pair
        bq.u[v] = *(const unsigned*)(wq_bf + h * EMB + e);   // B[e,h] = W[h,e]
        bk.u[v] = *(const unsigned*)(wk_bf + h * EMB + e);
      }
      accQ[nt] = wmma_bf16(a, bq, accQ[nt]);
      accK[nt] = wmma_bf16(a, bk, accK[nt]);
    }
  }
  // C layout: col n = lane&15, row m = v + 8*half
#pragma unroll
  for (int nt = 0; nt < 4; ++nt)
#pragma unroll
    for (int v = 0; v < 8; ++v) {
      const long row = rowBase + v + 8 * hf;
      const int  h   = nt * 16 + n;
      const unsigned short qb = f2bf1(accQ[nt][v]);
      q_bf[row * HD + h] = qb;
      k_bf[row * HD + h] = f2bf1(accK[nt][v]);
      // transposed copy of q (= v in the reference math)
      vT_bf[((long)b * HD + h) * SEQ + srow + v + 8 * hf] = qb;
    }
}

// ---------------------------------------------------------------------------
// Kernel 2: per-key-column softmax stats over the query axis.
// sT[k,q] = dot(k_row, q_row)/64. A = K rows (fixed per wave), B = Q^T.
// q-loop unrolled x2 with explicit double-buffered Q^T fragments so the next
// tile's loads are in flight while the current tile's WMMAs execute.
// ---------------------------------------------------------------------------
__global__ __launch_bounds__(256) void col_stats(
    const unsigned short* __restrict__ q_bf,
    const unsigned short* __restrict__ k_bf,
    float2* __restrict__ statOut)
{
  const int lane  = threadIdx.x & 31;
  const int wv    = threadIdx.x >> 5;
  const int b     = blockIdx.x >> 5;          // 32 blocks per batch
  const int kblk  = blockIdx.x & 31;
  const int ktile = kblk * 128 + wv * 16;
  const int n  = lane & 15;
  const int hf = lane >> 4;
  const long bbase = (long)b * SEQ;
  const float scale = 1.0f / 64.0f;           // 1/sqrt(SEQ)

  FragAB aK[2];                               // K rows, e in [0,32) and [32,64)
#pragma unroll
  for (int c = 0; c < 2; ++c)
#pragma unroll
    for (int v = 0; v < 8; ++v) {
      int kb = hf * 16 + v * 2;
      const long krow = bbase + ktile + n;
      aK[c].u[v] = *(const unsigned*)(k_bf + krow * HD + c * 32 + kb);
    }

  float ml[8], zl[8];
#pragma unroll
  for (int v = 0; v < 8; ++v) { ml[v] = -1e30f; zl[v] = 0.0f; }

  const int eb = hf * 16;                     // B-layout K base for this lane
  FragAB bA0, bA1, bB0, bB1;
  // preload tile qt=0
#pragma unroll
  for (int v = 0; v < 8; ++v) {
    const long qrow = bbase + n;
    bA0.u[v] = *(const unsigned*)(q_bf + qrow * HD + eb + v * 2);
    bA1.u[v] = *(const unsigned*)(q_bf + qrow * HD + 32 + eb + v * 2);
  }
  for (int qt = 0; qt < SEQ; qt += 32) {
    // issue loads for tile qt+16 (always exists: SEQ % 32 == 0)
    __builtin_prefetch(q_bf + (bbase + qt + 32 + n) * HD, 0, 1);
#pragma unroll
    for (int v = 0; v < 8; ++v) {
      const long qrow = bbase + qt + 16 + n;
      bB0.u[v] = *(const unsigned*)(q_bf + qrow * HD + eb + v * 2);
      bB1.u[v] = *(const unsigned*)(q_bf + qrow * HD + 32 + eb + v * 2);
    }
    v8f c = {};
    c = wmma_bf16(aK[0], bA0, c);
    c = wmma_bf16(aK[1], bA1, c);
#pragma unroll
    for (int v = 0; v < 8; ++v) {             // online max / sum-exp per row
      float s  = c[v] * scale;
      float nm = fmaxf(ml[v], s);
      zl[v] = zl[v] * __expf(ml[v] - nm) + __expf(s - nm);
      ml[v] = nm;
    }
    // issue loads for tile qt+32 while computing tile qt+16
    if (qt + 32 < SEQ) {
#pragma unroll
      for (int v = 0; v < 8; ++v) {
        const long qrow = bbase + qt + 32 + n;
        bA0.u[v] = *(const unsigned*)(q_bf + qrow * HD + eb + v * 2);
        bA1.u[v] = *(const unsigned*)(q_bf + qrow * HD + 32 + eb + v * 2);
      }
    }
    v8f d = {};
    d = wmma_bf16(aK[0], bB0, d);
    d = wmma_bf16(aK[1], bB1, d);
#pragma unroll
    for (int v = 0; v < 8; ++v) {
      float s  = d[v] * scale;
      float nm = fmaxf(ml[v], s);
      zl[v] = zl[v] * __expf(ml[v] - nm) + __expf(s - nm);
      ml[v] = nm;
    }
  }
  // merge across the 16 lanes holding the same row (xor 1,2,4,8 stays in-half)
#pragma unroll
  for (int v = 0; v < 8; ++v)
#pragma unroll
    for (int off = 1; off < 16; off <<= 1) {
      float om = __shfl_xor(ml[v], off, 32);
      float oz = __shfl_xor(zl[v], off, 32);
      float nm = fmaxf(ml[v], om);
      zl[v] = zl[v] * __expf(ml[v] - nm) + oz * __expf(om - nm);
      ml[v] = nm;
    }
  if (n == 0) {
#pragma unroll
    for (int v = 0; v < 8; ++v) {
      const long row = bbase + ktile + v + 8 * hf;
      statOut[row] = make_float2(ml[v], 1.0f / zl[v]);
    }
  }
}

// ---------------------------------------------------------------------------
// Kernel 3: out[q,h] = sum_k exp(s[q,k]-m_k)/Z_k * v[k,h], v = q_proj.
// All global loads for an iteration are issued up front; the V^T fragments
// are consumed only after score WMMAs + softmax + repack, hiding their latency.
// ---------------------------------------------------------------------------
__global__ __launch_bounds__(256) void attn_out(
    const unsigned short* __restrict__ q_bf,
    const unsigned short* __restrict__ k_bf,
    const unsigned short* __restrict__ vT_bf,
    const float2* __restrict__ statIn,
    float* __restrict__ out)
{
  __shared__ float2 sstat[SEQ];               // (m_k, 1/Z_k), 32 KiB
  const int lane  = threadIdx.x & 31;
  const int wv    = threadIdx.x >> 5;
  const int b     = blockIdx.x >> 5;
  const int qblk  = blockIdx.x & 31;
  const int qtile = qblk * 128 + wv * 16;
  const int n  = lane & 15;
  const int hf = lane >> 4;
  const long bbase = (long)b * SEQ;
  const float scale = 1.0f / 64.0f;

  for (int i = threadIdx.x; i < SEQ; i += 256)
    sstat[i] = statIn[bbase + i];
  __syncthreads();

  FragAB bq[2];                               // B = Q^T, fixed per wave
#pragma unroll
  for (int c = 0; c < 2; ++c)
#pragma unroll
    for (int v = 0; v < 8; ++v) {
      const long qrow = bbase + qtile + n;
      const int  e    = c * 32 + hf * 16 + v * 2;
      bq[c].u[v] = *(const unsigned*)(q_bf + qrow * HD + e);
    }

  v8f accO[4];
#pragma unroll
  for (int t = 0; t < 4; ++t) accO[t] = (v8f){};

  for (int kk = 0; kk < SEQ; kk += 32) {
    if (kk + 32 < SEQ) {                      // prefetch next K tiles (uniform)
      __builtin_prefetch(k_bf + (bbase + kk + 32 + n) * HD, 0, 1);
      __builtin_prefetch(vT_bf + ((long)b * HD + n) * SEQ + kk + 32, 0, 1);
    }
    // ---- issue ALL global loads for this iteration first ----
    FragAB aK00, aK01, aK10, aK11, aV[4];
#pragma unroll
    for (int v = 0; v < 8; ++v) {
      const int kb = hf * 16 + v * 2;
      const long r0 = (bbase + kk + n) * HD;
      const long r1 = (bbase + kk + 16 + n) * HD;
      aK00.u[v] = *(const unsigned*)(k_bf + r0 + kb);
      aK01.u[v] = *(const unsigned*)(k_bf + r0 + 32 + kb);
      aK10.u[v] = *(const unsigned*)(k_bf + r1 + kb);
      aK11.u[v] = *(const unsigned*)(k_bf + r1 + 32 + kb);
    }
#pragma unroll
    for (int ht = 0; ht < 4; ++ht) {
      const long vrow = ((long)b * HD + ht * 16 + n) * SEQ;
#pragma unroll
      for (int v = 0; v < 8; ++v) {
        const int kb = hf * 16 + v * 2;
        aV[ht].u[v] = *(const unsigned*)(vT_bf + vrow + kk + kb);
      }
    }
    // ---- score WMMAs: sT tiles (rows kk.. / kk+16..) ----
    v8f c0 = {}, c1 = {};
    c0 = wmma_bf16(aK00, bq[0], c0);
    c0 = wmma_bf16(aK01, bq[1], c0);
    c1 = wmma_bf16(aK10, bq[0], c1);
    c1 = wmma_bf16(aK11, bq[1], c1);
    // ---- softmax weights p = exp(s*scale - m_k) * (1/Z_k); row index = k ----
    float p0[8], p1[8];
#pragma unroll
    for (int v = 0; v < 8; ++v) {
      const float2 st0 = sstat[kk + v + 8 * hf];       // one ds_load_b64
      const float2 st1 = sstat[kk + 16 + v + 8 * hf];
      p0[v] = __expf(c0[v] * scale - st0.x) * st0.y;
      p1[v] = __expf(c1[v] * scale - st1.x) * st1.y;
    }
    // ---- repack P (two 16x16 C-layout tiles) -> one 32x16 B-layout frag ----
    // Target VGPR v, lane (n, bhalf=hf): rows 16*hf + {2v,2v+1}, col n.
    // Source lane = n + 16*(v>=4); tile index = hf.
    FragAB bp;
#pragma unroll
    for (int v = 0; v < 8; ++v) {
      const int j0 = (2 * v) & 7;
      const int j1 = j0 + 1;
      const int s  = (v >= 4) ? 1 : 0;
      float mine0 = hf ? p1[j0] : p0[j0];
      float mine1 = hf ? p1[j1] : p0[j1];
      float opp0  = hf ? p0[j0] : p1[j0];     // other lane's selector yields tile_hf
      float opp1  = hf ? p0[j1] : p1[j1];
      float sh0 = __shfl_xor(opp0, 16, 32);
      float sh1 = __shfl_xor(opp1, 16, 32);
      float r0 = (s == hf) ? mine0 : sh0;
      float r1 = (s == hf) ? mine1 : sh1;
      bp.u[v] = f2bf2(r0, r1);
    }
    // ---- out^T[h,q] += V^T[h, kk..kk+32) * P ; V = q_proj (reference bug) ----
#pragma unroll
    for (int ht = 0; ht < 4; ++ht)
      accO[ht] = wmma_bf16(aV[ht], bp, accO[ht]);
  }
  // store: out[(qtile+n), ht*16 + v + 8*hf]
#pragma unroll
  for (int ht = 0; ht < 4; ++ht)
#pragma unroll
    for (int v = 0; v < 8; ++v) {
      const long q = bbase + qtile + n;
      const int  h = ht * 16 + v + 8 * hf;
      out[q * HD + h] = accO[ht][v];
    }
}

// ---------------------------------------------------------------------------
extern "C" void kernel_launch(void* const* d_in, const int* in_sizes, int n_in,
                              void* d_out, int out_size, void* d_ws, size_t ws_size,
                              hipStream_t stream) {
  const float* x  = (const float*)d_in[0];   // [8,4096,768]
  const float* Wq = (const float*)d_in[1];   // [64,768]
  const float* Wk = (const float*)d_in[2];   // [64,768]
  // d_in[3] = Wv: unused by the reference math (source bug uses q_net for v)
  float* out = (float*)d_out;                // [8,4096,64] f32

  char* ws = (char*)d_ws;
  unsigned short* q_bf  = (unsigned short*)(ws);                   // 4 MiB
  unsigned short* k_bf  = (unsigned short*)(ws + 4194304);         // 4 MiB
  unsigned short* vT_bf = (unsigned short*)(ws + 8388608);         // 4 MiB
  float2*         stat  = (float2*)(ws + 12582912);                // 256 KiB
  unsigned short* wqb   = (unsigned short*)(ws + 12845056);        // 96 KiB
  unsigned short* wkb   = (unsigned short*)(ws + 12943360);        // 96 KiB

  wconvert<<<(HD * EMB + 255) / 256, 256, 0, stream>>>(Wq, Wk, wqb, wkb);
  proj_qk<<<256, 256, 0, stream>>>(x, wqb, wkb, q_bf, k_bf, vT_bf);
  col_stats<<<256, 256, 0, stream>>>(q_bf, k_bf, stat);
  attn_out<<<256, 256, 0, stream>>>(q_bf, k_bf, vT_bf, stat, out);
}
